// DeMultiheadAttention_74809740362221
// MI455X (gfx1250) — matrix operand order repaired and checked by reading, not compiled, not run on hardware
//
#include <hip/hip_runtime.h>
#include <cmath>

typedef __attribute__((ext_vector_type(16))) _Float16 v16h;
typedef __attribute__((ext_vector_type(8)))  float    v8f;

#define WMMA_F16(a, b, c) \
  __builtin_amdgcn_wmma_f32_16x16x32_f16(false, (a), false, (b), (short)0, (c), false, false)

namespace {
constexpr int kD  = 768;
constexpr int kH  = 12;
constexpr int kHd = 64;
constexpr int kB  = 8;
constexpr int kL  = 1024;
constexpr int kM  = kB * kL;  // 8192 rows
}

union FragA { v16h v; _Float16 h[16]; uint4 q[2]; };
union Acc8  { v8f  v; float    f[8]; };

__device__ __forceinline__ void cvt4(_Float16* h, float4 f) {
  h[0] = (_Float16)f.x; h[1] = (_Float16)f.y;
  h[2] = (_Float16)f.z; h[3] = (_Float16)f.w;
}

// B-fragment: 16 contiguous halves (K = hsel*16 + 0..15 within the 32-K step).
__device__ __forceinline__ v16h load_bfrag(const _Float16* p) {
  union { v16h v; uint4 q[2]; } u;
  const uint4* s = (const uint4*)p;
  u.q[0] = s[0];
  u.q[1] = s[1];
  return u.v;
}

// ---------------------------------------------------------------------------
// One-shot f32 -> f16 conversion (x, pos, weights). 8 elements / thread.
// ---------------------------------------------------------------------------
__global__ __launch_bounds__(256) void cvt_f32_to_f16(
    const float* __restrict__ in, _Float16* __restrict__ out, int n)
{
  const int i = (blockIdx.x * 256 + threadIdx.x) * 8;
  if (i + 8 <= n) {
    const float4 a = *(const float4*)(in + i);
    const float4 b = *(const float4*)(in + i + 4);
    union { uint4 u; _Float16 h[8]; } p;
    cvt4(p.h, a);
    cvt4(p.h + 4, b);
    *(uint4*)(out + i) = p.u;
  }
}

// ---------------------------------------------------------------------------
// Projection GEMM: C = A(Mx768) @ W(Nx768)^T + bias, f16 in / f32 accum.
//   mode 0: N=2304, cols [0,768)->Q [768,1536)->K [1536,2304)->V(transposed)
//   mode 1: N=768,  all cols -> outS, scaled by out_scale (pk gets 1/48)
// Block = 128 threads (4 waves); block tile 128(M) x 64(N);
// wave tile 32(M) x 64(N): each B fragment feeds 2 WMMAs (1.5 loads/WMMA).
// K-loop double-buffered: pure global_load_b128 + v_wmma.
// ---------------------------------------------------------------------------
__global__ __launch_bounds__(128, 1) void proj_gemm_f16(
    const _Float16* __restrict__ A, const _Float16* __restrict__ W,
    const float* __restrict__ bias, int mode, float out_scale,
    _Float16* __restrict__ outQ, _Float16* __restrict__ outK,
    _Float16* __restrict__ outV, _Float16* __restrict__ outS)
{
  const int lane = threadIdx.x & 31;
  const int wave = threadIdx.x >> 5;
  const int mrow = lane & 15;   // A row within 16-row tile / B column
  const int hsel = lane >> 4;   // lane-half selector
  const int m0 = blockIdx.x * 128 + wave * 32;
  const int n0 = blockIdx.y * 64;

  Acc8 acc[2][4];
#pragma unroll
  for (int mh = 0; mh < 2; ++mh)
#pragma unroll
    for (int g = 0; g < 4; ++g)
#pragma unroll
      for (int r = 0; r < 8; ++r) acc[mh][g].f[r] = 0.0f;

  const _Float16* arow0 = A + (size_t)(m0 + mrow) * kD;
  const _Float16* arow1 = A + (size_t)(m0 + 16 + mrow) * kD;
  const _Float16* wbase = W + (size_t)n0 * kD;

  FragA af[2][2];
  FragA bf[2][4];

  auto load_step = [&](int buf, int k0) {
    // A fragments: elems 0..15 -> K = k0+hsel*16+(0..15)
    const _Float16* ap0 = arow0 + k0 + hsel * 16;
    af[buf][0].q[0] = *(const uint4*)ap0;
    af[buf][0].q[1] = *(const uint4*)(ap0 + 8);
    const _Float16* ap1 = arow1 + k0 + hsel * 16;
    af[buf][1].q[0] = *(const uint4*)ap1;
    af[buf][1].q[1] = *(const uint4*)(ap1 + 8);
#pragma unroll
    for (int g = 0; g < 4; ++g) {
      // B fragment: lane = col n, 16 contiguous K halves from W[n][k]
      const _Float16* wp = wbase + (size_t)(g * 16 + mrow) * kD + k0 + hsel * 16;
      bf[buf][g].q[0] = *(const uint4*)wp;
      bf[buf][g].q[1] = *(const uint4*)(wp + 8);
    }
  };

  load_step(0, 0);
  int buf = 0;
  for (int k0 = 0; k0 < kD; k0 += 32) {
    if (k0 + 32 < kD) load_step(buf ^ 1, k0 + 32);  // loads overlap the WMMAs
#pragma unroll
    for (int g = 0; g < 4; ++g) {
      acc[0][g].v = WMMA_F16(af[buf][0].v, bf[buf][g].v, acc[0][g].v);
      acc[1][g].v = WMMA_F16(af[buf][1].v, bf[buf][g].v, acc[1][g].v);
    }
    buf ^= 1;
  }

  const int b     = m0 / kL;    // 32-row tile never crosses a batch boundary
  const int lbase = m0 % kL;

#pragma unroll
  for (int mh = 0; mh < 2; ++mh) {
    const int lb = lbase + mh * 16;
#pragma unroll
    for (int g = 0; g < 4; ++g) {
      const int n   = n0 + g * 16 + mrow;   // this lane's output channel
      const float bv = bias[n];
      const int seg = n / kD;
      const int hh  = (n % kD) >> 6;
      const int d   = n & 63;
      if (mode == 0 && seg == 2) {
        // V: store transposed [b,h,d,l]; thread holds 8 consecutive l values.
        union { uint4 u; _Float16 h[8]; } pk;
#pragma unroll
        for (int r = 0; r < 8; ++r) pk.h[r] = (_Float16)(acc[mh][g].f[r] + bv);
        _Float16* dst = outV + ((size_t)(b * kH + hh) * kHd + d) * kL + (lb + 8 * hsel);
        *(uint4*)dst = pk.u;
      } else {
        _Float16* base;
        float sc = out_scale;
        if (mode == 0) { base = (seg == 0) ? outQ : outK; sc = 1.0f; }
        else           { base = outS; }
        base += (size_t)(b * kH + hh) * kL * kHd + d;
#pragma unroll
        for (int r = 0; r < 8; ++r) {
          const int l = lb + 8 * hsel + r;   // C layout: row = r + 8*hsel
          base[(size_t)l * kHd] = (_Float16)((acc[mh][g].f[r] + bv) * sc);
        }
      }
    }
  }
}

// ---------------------------------------------------------------------------
// Fused disentangled attention with online (flash) softmax.
//   S = Q K^T + Q PQ^T + K PK'^T   (PK' pre-scaled by 1/sqrt(3d))
//   out = softmax(S) V
// Block = 128 threads (4 waves); each block: one (b,h), 64 query rows;
// each wave: 16 query rows, loops over 16 key chunks of 64.
// ---------------------------------------------------------------------------
__global__ __launch_bounds__(128, 1) void deattn_kernel(
    const _Float16* __restrict__ Qf, const _Float16* __restrict__ Kf,
    const _Float16* __restrict__ Vt, const _Float16* __restrict__ PQf,
    const _Float16* __restrict__ PKf, float* __restrict__ out)
{
  __shared__ __align__(16) _Float16 plds[4][16 * 64];  // per-wave P staging

  const int lane = threadIdx.x & 31;
  const int wave = threadIdx.x >> 5;
  const int mrow = lane & 15;
  const int hsel = lane >> 4;
  const int bh = blockIdx.y;
  const int b = bh / kH, h = bh % kH;
  const int qbase = blockIdx.x * 64 + wave * 16;

  const size_t bhoff = (size_t)bh * kL * kHd;
  const _Float16* Qb  = Qf  + bhoff;
  const _Float16* Kb  = Kf  + bhoff;
  const _Float16* PQb = PQf + bhoff;
  const _Float16* PKb = PKf + bhoff;
  const _Float16* Vb  = Vt  + bhoff;  // [64][1024]

  // Preload A-operand row fragments for q and k rows (hd = 64 -> 2 K-steps each).
  FragA qf[2], kf[2];
#pragma unroll
  for (int j = 0; j < 2; ++j) {
    const _Float16* qp = Qb + (size_t)(qbase + mrow) * kHd + j * 32 + hsel * 16;
    qf[j].q[0] = *(const uint4*)qp;
    qf[j].q[1] = *(const uint4*)(qp + 8);
    const _Float16* kp = Kb + (size_t)(qbase + mrow) * kHd + j * 32 + hsel * 16;
    kf[j].q[0] = *(const uint4*)kp;
    kf[j].q[1] = *(const uint4*)(kp + 8);
  }

  Acc8 O[4];
  float mstate[8], lstate[8];
#pragma unroll
  for (int g = 0; g < 4; ++g)
#pragma unroll
    for (int r = 0; r < 8; ++r) O[g].f[r] = 0.0f;
#pragma unroll
  for (int r = 0; r < 8; ++r) { mstate[r] = -3.0e38f; lstate[r] = 0.0f; }

  _Float16* pw = plds[wave];

  for (int kc = 0; kc < 16; ++kc) {
    const int l0 = kc * 64;

    // ---- S = Q*K^T + Q*PQ^T + K*PK'^T over this 64-key chunk ----
    Acc8 S[4];
#pragma unroll
    for (int g = 0; g < 4; ++g)
#pragma unroll
      for (int r = 0; r < 8; ++r) S[g].f[r] = 0.0f;

#pragma unroll
    for (int g = 0; g < 4; ++g) {
      const size_t lrow = (size_t)(l0 + g * 16 + mrow) * kHd;
#pragma unroll
      for (int j = 0; j < 2; ++j) {
        const int off = j * 32 + hsel * 16;
        v16h bk = load_bfrag(Kb + lrow + off);
        S[g].v = WMMA_F16(qf[j].v, bk, S[g].v);
        v16h bq = load_bfrag(PQb + lrow + off);
        S[g].v = WMMA_F16(qf[j].v, bq, S[g].v);
        v16h bp = load_bfrag(PKb + lrow + off);
        S[g].v = WMMA_F16(kf[j].v, bp, S[g].v);
      }
    }

    if (kc + 1 < 16) {  // warm next key chunk (global_prefetch_b8)
      const size_t nl = (size_t)(l0 + 64 + lane) * kHd;
      __builtin_prefetch(Kb + nl, 0, 1);
      __builtin_prefetch(PQb + nl, 0, 1);
      __builtin_prefetch(PKb + nl, 0, 1);
    }

    // ---- online softmax (rows live in 16-lane halves of C layout) ----
    float rmax[8];
#pragma unroll
    for (int r = 0; r < 8; ++r) {
      float t = fmaxf(fmaxf(S[0].f[r], S[1].f[r]), fmaxf(S[2].f[r], S[3].f[r]));
#pragma unroll
      for (int off = 1; off < 16; off <<= 1) t = fmaxf(t, __shfl_xor(t, off, 32));
      rmax[r] = t;
    }
#pragma unroll
    for (int r = 0; r < 8; ++r) {
      const float mn = fmaxf(mstate[r], rmax[r]);
      const float alpha = __expf(mstate[r] - mn);
      mstate[r] = mn;
      float rs = 0.0f;
#pragma unroll
      for (int g = 0; g < 4; ++g) {
        const float p = __expf(S[g].f[r] - mn);
        S[g].f[r] = p;
        rs += p;
      }
#pragma unroll
      for (int off = 1; off < 16; off <<= 1) rs += __shfl_xor(rs, off, 32);
      lstate[r] = lstate[r] * alpha + rs;
#pragma unroll
      for (int g = 0; g < 4; ++g) O[g].f[r] *= alpha;
    }

    // ---- reshape P (C layout -> A fragments) via per-wave LDS ----
#pragma unroll
    for (int g = 0; g < 4; ++g)
#pragma unroll
      for (int r = 0; r < 8; ++r)
        pw[(r + 8 * hsel) * 64 + g * 16 + mrow] = (_Float16)S[g].f[r];

    asm volatile("s_wait_dscnt 0" ::: "memory");  // intra-wave LDS RAW guard

    FragA pf[2];
#pragma unroll
    for (int j = 0; j < 2; ++j) {
      const _Float16* pp = pw + mrow * 64 + j * 32 + hsel * 16;
      pf[j].q[0] = *(const uint4*)pp;
      pf[j].q[1] = *(const uint4*)(pp + 8);
    }

    // ---- O += P @ V  (V transposed [d][l] -> contiguous B fragments) ----
#pragma unroll
    for (int dg = 0; dg < 4; ++dg) {
      const _Float16* vrow = Vb + (size_t)(dg * 16 + mrow) * kL + l0;
#pragma unroll
      for (int j = 0; j < 2; ++j) {
        v16h bv = load_bfrag(vrow + j * 32 + hsel * 16);
        O[dg].v = WMMA_F16(pf[j].v, bv, O[dg].v);
      }
    }
  }

  // ---- normalize and store: out (b, n, h, hd) f32 ----
#pragma unroll
  for (int dg = 0; dg < 4; ++dg)
#pragma unroll
    for (int r = 0; r < 8; ++r) {
      const int l = qbase + 8 * hsel + r;
      const int d = dg * 16 + mrow;
      out[(((size_t)b * kL + l) * kH + h) * kHd + d] = O[dg].f[r] / lstate[r];
    }
}

// ---------------------------------------------------------------------------
extern "C" void kernel_launch(void* const* d_in, const int* in_sizes, int n_in,
                              void* d_out, int out_size, void* d_ws, size_t ws_size,
                              hipStream_t stream) {
  (void)in_sizes; (void)n_in; (void)out_size; (void)ws_size;
  const float* x     = (const float*)d_in[0];
  const float* pos   = (const float*)d_in[1];
  const float* qkv_w = (const float*)d_in[2];
  const float* qkv_b = (const float*)d_in[3];
  const float* pq_w  = (const float*)d_in[4];
  const float* pq_b  = (const float*)d_in[5];
  const float* pk_w  = (const float*)d_in[6];
  const float* pk_b  = (const float*)d_in[7];
  float* outp = (float*)d_out;

  const size_t tsz = (size_t)kB * kH * kL * kHd;  // 6.29M halves per tensor
  _Float16* Qf   = (_Float16*)d_ws;
  _Float16* Kf   = Qf + tsz;
  _Float16* Vt   = Kf + tsz;
  _Float16* PQf  = Vt + tsz;
  _Float16* PKf  = PQf + tsz;
  _Float16* Xh   = PKf + tsz;                     // (8192, 768) f16
  _Float16* Ph   = Xh + (size_t)kM * kD;          // (8192, 768) f16
  _Float16* Wqkv = Ph + (size_t)kM * kD;          // (2304, 768) f16
  _Float16* Wpq  = Wqkv + (size_t)3 * kD * kD;    // (768, 768) f16
  _Float16* Wpk  = Wpq + (size_t)kD * kD;         // (768, 768) f16

  const float inv_scale = 1.0f / sqrtf(3.0f * (float)kD);  // 1/48

  // ---- pre-convert activations + weights to f16 (one pass each) ----
  const int nAct = kM * kD;           // 6,291,456
  const int nWq  = 3 * kD * kD;       // 1,769,472
  const int nWp  = kD * kD;           //   589,824
  cvt_f32_to_f16<<<dim3(nAct / 2048), dim3(256), 0, stream>>>(x, Xh, nAct);
  cvt_f32_to_f16<<<dim3(nAct / 2048), dim3(256), 0, stream>>>(pos, Ph, nAct);
  cvt_f32_to_f16<<<dim3(nWq / 2048), dim3(256), 0, stream>>>(qkv_w, Wqkv, nWq);
  cvt_f32_to_f16<<<dim3(nWp / 2048), dim3(256), 0, stream>>>(pq_w, Wpq, nWp);
  cvt_f32_to_f16<<<dim3(nWp / 2048), dim3(256), 0, stream>>>(pk_w, Wpk, nWp);

  // ---- projections (WMMA f16) ----
  dim3 blk(128);
  proj_gemm_f16<<<dim3(kM / 128, (3 * kD) / 64), blk, 0, stream>>>(
      Xh, Wqkv, qkv_b, 0, 1.0f, Qf, Kf, Vt, nullptr);
  proj_gemm_f16<<<dim3(kM / 128, kD / 64), blk, 0, stream>>>(
      Ph, Wpq, pq_b, 1, 1.0f, nullptr, nullptr, nullptr, PQf);
  proj_gemm_f16<<<dim3(kM / 128, kD / 64), blk, 0, stream>>>(
      Ph, Wpk, pk_b, 1, inv_scale, nullptr, nullptr, nullptr, PKf);

  // ---- fused attention (WMMA f16 + online softmax) ----
  deattn_kernel<<<dim3(kL / 64, kB * kH), blk, 0, stream>>>(
      Qf, Kf, Vt, PQf, PKf, outp);
}
